// LearnedDataFilterBase_82188494177069
// MI455X (gfx1250) — matrix operand order repaired and checked by reading, not compiled, not run on hardware
//
#include <hip/hip_runtime.h>

// ---------------------------------------------------------------------------
// LearnedDataFilter for MI455X (gfx1250, wave32, WMMA + TDM async staging)
//
//   k0: pack w1 (fp32 [128][512][3]) into exact WMMA B-fragments (f16)
//   k1: conv1 (stride 3, K=3, pad 1) as WMMA GEMM + bias + maxout -> hm
//       - B fragments DMA'd to LDS per k-chunk via Tensor Data Mover,
//         double buffered (cooperative ds_store fallback)
//       - A operand software-pipelined from global (read-once stream)
//       - B fragments staged LDS->regs in groups of 4 to cluster ds_loads
//   k2: conv2 (64->1, stride 3) + per-batch score est[b] = sum(s)/sl
//   k3: top-32 of 64 scores (stable tie-break), sorted indices, sel_lens
//   k4: gather copy x[idx][:, :max_time, :] with non-temporal b128 traffic
// ---------------------------------------------------------------------------

typedef __attribute__((ext_vector_type(16))) _Float16     v16h;
typedef __attribute__((ext_vector_type(8)))  float        v8f;
typedef __attribute__((ext_vector_type(4)))  float        f32x4;
typedef __attribute__((ext_vector_type(4)))  unsigned int u32x4;
typedef __attribute__((ext_vector_type(8)))  int          i32x8;
typedef __attribute__((ext_vector_type(4)))  int          i32x4;

#define B_   64
#define T_   2048
#define D_   512
#define H_   64
#define K_   3
#define T1_  683          // conv1 output length: (2048 + 2 - 3)/3 + 1
#define T1P_ 688          // padded rows in hm
#define T2_  228          // conv2 output length: (683 + 2 - 3)/3 + 1
#define NKCH 48           // 3 taps * (512/32) d-chunks
#define NFRAG (NKCH * 8)  // kchunks * ntiles(128/16)
#define CHUNK_HALFS 4096  // 8 fragments * 512 halfs = 8KB per k-chunk

#define WF_BYTES ((size_t)NFRAG * 512 * 2)            // 393216 B packed weights
#define HM_BYTES ((size_t)B_ * T1P_ * H_ * 4)         // ~11.3 MB

#define HAVE_TDM __has_builtin(__builtin_amdgcn_tensor_load_to_lds)

// --- k0: pack w1 into per-(kchunk,ntile) B fragments -----------------------
// fragment f = kc*8 + n. Lane L, element j:  K = j + 16*(L>=16), N = L%16
//   -> d = dbase + K,  c = n*16 + N,  value = w1[c][d][tap]
__global__ void ldf_pack_w(const float* __restrict__ w1, _Float16* __restrict__ wf) {
    int tid  = blockIdx.x * 256 + threadIdx.x;
    int frag = tid >> 5;
    int lane = tid & 31;
    if (frag >= NFRAG) return;
    int kc    = frag >> 3;
    int n     = frag & 7;
    int tap   = kc >> 4;
    int dbase = (kc & 15) << 5;
    int c     = (n << 4) + (lane & 15);
    int d0    = dbase + ((lane >> 4) << 4);   // +16 for lanes 16..31
    _Float16* dst = wf + ((size_t)frag << 9) + (lane << 4);
#pragma unroll
    for (int j = 0; j < 16; ++j) {
        float v = w1[(size_t)c * (D_ * K_) + (size_t)(d0 + j) * K_ + tap];
        dst[j] = (_Float16)v;
    }
}

#if HAVE_TDM
// Issue a TDM DMA of one 8KB B k-chunk (1024 x 8-byte elements, 1D tile)
// from global `gsrc` into LDS byte offset `lds_off`.
__device__ __forceinline__ void tdm_load_b_chunk(const _Float16* gsrc,
                                                 unsigned lds_off) {
    unsigned long long ga = (unsigned long long)(const void*)gsrc;
    u32x4 g0;
    g0[0] = 1u;                                            // count=1, user D#
    g0[1] = lds_off;                                       // lds_addr (bytes)
    g0[2] = (unsigned)(ga & 0xFFFFFFFFu);                  // global_addr lo
    g0[3] = (unsigned)((ga >> 32) & 0x1FFFFFFu) | (2u << 30); // addr hi | type=2
    i32x8 g1;
    g1[0] = (int)(3u << 16);        // workgroup_mask=0, data_size=3 (8B)
    g1[1] = (int)(1024u << 16);     // tensor_dim0 = 1024 (bits [79:48] lo16)
    g1[2] = (int)(1u << 16);        // tensor_dim0 hi = 0, tensor_dim1 = 1
    g1[3] = (int)(1024u << 16);     // tensor_dim1 hi = 0, tile_dim0 = 1024
    g1[4] = 0;                      // tile_dim1 = 0 (unused), tile_dim2 = 0
    g1[5] = 1024;                   // tensor_dim0_stride = 1024 elements
    g1[6] = 0;
    g1[7] = 0;
    i32x4 z4 = {0, 0, 0, 0};
#if __clang_major__ >= 23
    i32x8 z8 = {0, 0, 0, 0, 0, 0, 0, 0};
    __builtin_amdgcn_tensor_load_to_lds(g0, g1, z4, z4, z8, 0);
#else
    __builtin_amdgcn_tensor_load_to_lds(g0, g1, z4, z4, 0);
#endif
}
#endif

#define WMMA_F16(A, BF, C) \
    __builtin_amdgcn_wmma_f32_16x16x32_f16(false, (A), false, (BF), (short)0, (C), false, false)

// --- k1: conv1 as WMMA GEMM + bias + maxout --------------------------------
// grid = (6 time-tiles, 64 batches), 256 threads = 8 waves.
// Wave w owns an M-tile of 16 output times; N = all 128 channels (8 frags).
__global__ void __launch_bounds__(256)
ldf_conv1_wmma(const float* __restrict__ x, const _Float16* __restrict__ wf,
               const float* __restrict__ b1, float* __restrict__ hm) {
    __shared__ _Float16 sB[2][CHUNK_HALFS];   // double-buffered 8KB B chunks

    const int b        = blockIdx.y;
    const int tid      = threadIdx.x;
    const int lane     = tid & 31;
    const int half_sel = lane >> 4;          // lanes 16..31
    const int m        = lane & 15;
    const int t0       = blockIdx.x * 128 + (tid >> 5) * 16;
    const int t        = t0 + m;             // A-matrix row (output time)

    v8f acc[8] = {};

    // A operand: raw fp32 for the *current* k-chunk, software pipelined.
    float4 a0, a1, a2, a3;
    auto loadA = [&](int kc, float4& u0, float4& u1, float4& u2, float4& u3) {
        const int tap   = kc >> 4;
        const int dbase = (kc & 15) << 5;
        const int time  = 3 * t - 1 + tap;   // input time (pad=1)
        const bool ok   = (time >= 0) && (time < T_);
        const float* px = x + ((size_t)b * T_ + (size_t)(ok ? time : 0)) * D_
                            + dbase + (half_sel << 4);
        if (ok) {
            u0 = *(const float4*)(px);
            u1 = *(const float4*)(px + 4);
            u2 = *(const float4*)(px + 8);
            u3 = *(const float4*)(px + 12);
        } else {
            u0 = u1 = u2 = u3 = make_float4(0.f, 0.f, 0.f, 0.f);
        }
    };

    // ---- B staging prologue --------------------------------------------
#if HAVE_TDM
    const unsigned lds_base = (unsigned)(size_t)(void*)&sB[0][0];
    if (tid < 32) tdm_load_b_chunk(wf, lds_base);          // chunk 0 -> buf 0
#else
    float4 nb0, nb1;                                        // prefetch regs
    {   // stage chunk 0 directly (each thread moves 32B; 256*32B = 8KB)
        const float4* src = (const float4*)(wf) + tid * 2;
        float4 p0 = src[0], p1 = src[1];
        float4* dst = (float4*)&sB[0][0] + tid * 2;
        dst[0] = p0; dst[1] = p1;
        const float4* s1 = (const float4*)(wf + CHUNK_HALFS) + tid * 2;
        nb0 = s1[0]; nb1 = s1[1];                           // prefetch chunk 1
    }
    __syncthreads();
#endif
    loadA(0, a0, a1, a2, a3);

    for (int kc = 0; kc < NKCH; ++kc) {
        const int cur = kc & 1;
#if HAVE_TDM
        if (tid < 32) __builtin_amdgcn_s_wait_tensorcnt(0); // chunk kc landed
        __syncthreads();            // publish buf[cur]; all reads of other buf done
        if (kc + 1 < NKCH && tid < 32)
            tdm_load_b_chunk(wf + (size_t)(kc + 1) * CHUNK_HALFS,
                             lds_base + (unsigned)((1 - cur) * 8192));
#endif
        // Convert A for this chunk; then prefetch next chunk's raw data.
        // A fragment (16x32 f16): elem j -> K=j+16*half_sel
        v16h a;
        a[0]  = (_Float16)a0.x; a[1]  = (_Float16)a0.y;
        a[2]  = (_Float16)a0.z; a[3]  = (_Float16)a0.w;
        a[4]  = (_Float16)a1.x; a[5]  = (_Float16)a1.y;
        a[6]  = (_Float16)a1.z; a[7]  = (_Float16)a1.w;
        a[8]  = (_Float16)a2.x; a[9]  = (_Float16)a2.y;
        a[10] = (_Float16)a2.z; a[11] = (_Float16)a2.w;
        a[12] = (_Float16)a3.x; a[13] = (_Float16)a3.y;
        a[14] = (_Float16)a3.z; a[15] = (_Float16)a3.w;
        if (kc + 1 < NKCH) loadA(kc + 1, a0, a1, a2, a3);

        // B fragments staged LDS->regs in groups of 4: 8 clustered
        // ds_load_b128, one dscnt wait, 4 back-to-back WMMAs per group.
        const _Float16* bbase = &sB[cur][lane << 4];
        {
            v16h bf0 = *(const v16h*)(bbase);
            v16h bf1 = *(const v16h*)(bbase + 512);
            v16h bf2 = *(const v16h*)(bbase + 1024);
            v16h bf3 = *(const v16h*)(bbase + 1536);
            acc[0] = WMMA_F16(a, bf0, acc[0]);
            acc[1] = WMMA_F16(a, bf1, acc[1]);
            acc[2] = WMMA_F16(a, bf2, acc[2]);
            acc[3] = WMMA_F16(a, bf3, acc[3]);
        }
        {
            v16h bf4 = *(const v16h*)(bbase + 2048);
            v16h bf5 = *(const v16h*)(bbase + 2560);
            v16h bf6 = *(const v16h*)(bbase + 3072);
            v16h bf7 = *(const v16h*)(bbase + 3584);
            acc[4] = WMMA_F16(a, bf4, acc[4]);
            acc[5] = WMMA_F16(a, bf5, acc[5]);
            acc[6] = WMMA_F16(a, bf6, acc[6]);
            acc[7] = WMMA_F16(a, bf7, acc[7]);
        }

#if !HAVE_TDM
        if (kc + 1 < NKCH) {
            __syncthreads();        // everyone done reading buf[1-cur]
            float4* dst = (float4*)&sB[1 - cur][0] + tid * 2;
            dst[0] = nb0; dst[1] = nb1;
            if (kc + 2 < NKCH) {
                const float4* src =
                    (const float4*)(wf + (size_t)(kc + 2) * CHUNK_HALFS) + tid * 2;
                nb0 = src[0]; nb1 = src[1];
            }
            __syncthreads();        // new buffer published
        }
#endif
    }

    // Epilogue: bias + maxout across adjacent channels (adjacent N lanes).
    // D-frag: elem r at lane L -> M = r + 8*(L>=16), N = L%16.
#pragma unroll
    for (int n = 0; n < 8; ++n) {
        const float bias = b1[(n << 4) + m];
#pragma unroll
        for (int r = 0; r < 8; ++r) {
            float v  = acc[n][r] + bias;
            float o  = __shfl_xor(v, 1, 32);
            float mx = fmaxf(v, o);
            int tt   = t0 + r + (half_sel << 3);
            if (((lane & 1) == 0) && tt < T1_) {
                int cout = (n << 3) + (m >> 1);
                hm[((size_t)b * T1P_ + tt) * H_ + cout] = mx;
            }
        }
    }
}

// --- k2: conv2 + score: est[b] = sum_t2 s[t2] / sl(b) ----------------------
__global__ void __launch_bounds__(256)
ldf_conv2_est(const float* __restrict__ hm, const float* __restrict__ w2,
              const float* __restrict__ b2, const int* __restrict__ seq_lens,
              float* __restrict__ est) {
    __shared__ float sw2[3 * H_];
    __shared__ float red[256];
    const int b   = blockIdx.x;
    const int tid = threadIdx.x;
    if (tid < 3 * H_) {
        int k = tid / H_, c = tid % H_;
        sw2[k * H_ + c] = w2[c * K_ + k];
    }
    __syncthreads();
    float s = 0.f;
    if (tid < T2_) {
        float accv = b2[0];
#pragma unroll
        for (int k = 0; k < 3; ++k) {
            int t1 = 3 * tid - 1 + k;
            if (t1 >= 0 && t1 < T1_) {
                const float* row = hm + ((size_t)b * T1P_ + t1) * H_;
                const float* wk  = sw2 + k * H_;
#pragma unroll
                for (int c4 = 0; c4 < H_ / 4; ++c4) {
                    float4 av = *(const float4*)(row + 4 * c4);
                    float4 wv = *(const float4*)(wk + 4 * c4);
                    accv += av.x * wv.x + av.y * wv.y + av.z * wv.z + av.w * wv.w;
                }
            }
        }
        s = accv;
    }
    red[tid] = s;
    __syncthreads();
    for (int off = 128; off > 0; off >>= 1) {
        if (tid < off) red[tid] += red[tid + off];
        __syncthreads();
    }
    if (tid == 0) {
        int L   = seq_lens[b];
        int sl1 = (L + 2) / 3;
        int sl2 = (sl1 + 2) / 3;
        est[b]  = red[0] / (float)sl2;   // mean * (T2/sl) == sum/sl
    }
}

// --- k3: top-32 of 64, sorted indices; write sel_lens ----------------------
__global__ void ldf_topk(const float* __restrict__ est,
                         const int* __restrict__ seq_lens,
                         int* __restrict__ sel_idx,
                         float* __restrict__ out_lens) {
    __shared__ float e[B_];
    __shared__ int   flag[B_];
    const int i = threadIdx.x;
    e[i] = est[i];
    __syncthreads();
    const float ei = e[i];
    int r = 0;
    for (int j = 0; j < B_; ++j) {
        float ej = e[j];
        if (ej > ei || (ej == ei && j < i)) ++r;   // jax top_k tie-break
    }
    int sel = (r < B_ / 2) ? 1 : 0;
    flag[i] = sel;
    __syncthreads();
    if (sel) {
        int pos = 0;
        for (int j = 0; j < i; ++j) pos += flag[j];
        sel_idx[pos]  = i;
        out_lens[pos] = (float)seq_lens[i];
    }
}

// --- k4: gather x[idx][:, :max_time, :], non-temporal both ways ------------
// (write-once stream that can't live in L2; keep it from evicting hot data)
// Uses plain ext-vector f32x4 because the nontemporal builtins reject the
// HIP_vector_type struct.
__global__ void __launch_bounds__(256)
ldf_gather(const float* __restrict__ x, const int* __restrict__ sel_idx,
           float* __restrict__ out, int cnt4) {
    const int row = blockIdx.y;
    const int g   = blockIdx.x * 256 + threadIdx.x;
    if (g >= cnt4) return;
    const int src_row = sel_idx[row];
    const f32x4* src = (const f32x4*)x + (size_t)src_row * (T_ * D_ / 4) + g;
    f32x4*       dst = (f32x4*)out + (size_t)row * cnt4 + g;
    f32x4 v = __builtin_nontemporal_load(src);
    __builtin_nontemporal_store(v, dst);
}

extern "C" void kernel_launch(void* const* d_in, const int* in_sizes, int n_in,
                              void* d_out, int out_size, void* d_ws, size_t ws_size,
                              hipStream_t stream) {
    const float* x        = (const float*)d_in[0];
    const int*   seq_lens = (const int*)d_in[1];
    const float* w1       = (const float*)d_in[2];
    const float* b1       = (const float*)d_in[3];
    const float* w2       = (const float*)d_in[4];
    const float* b2       = (const float*)d_in[5];
    float*       out      = (float*)d_out;

    char* ws = (char*)d_ws;
    _Float16* wf      = (_Float16*)ws;
    float*    hm      = (float*)(ws + WF_BYTES);
    float*    est     = (float*)(ws + WF_BYTES + HM_BYTES);
    int*      sel_idx = (int*)(ws + WF_BYTES + HM_BYTES + 256);

    // Output = [x_sel (32 x max_time x 512) | sel_lens (32)]
    int max_time = (out_size - 32) / (32 * 512);
    if (max_time < 1) max_time = 1;

    ldf_pack_w<<<(NFRAG * 32 + 255) / 256, 256, 0, stream>>>(w1, wf);
    ldf_conv1_wmma<<<dim3(6, B_), 256, 0, stream>>>(x, wf, b1, hm);
    ldf_conv2_est<<<B_, 256, 0, stream>>>(hm, w2, b2, seq_lens, est);
    ldf_topk<<<1, B_, 0, stream>>>(est, seq_lens, sel_idx,
                                   out + (size_t)32 * max_time * 512);
    int cnt4 = max_time * (D_ / 4);   // float4s per selected row
    ldf_gather<<<dim3((cnt4 + 255) / 256, 32), 256, 0, stream>>>(x, sel_idx, out, cnt4);
}
